// CellVGAE_GCNEncoder_37495064494308
// MI455X (gfx1250) — matrix operand order repaired and checked by reading, not compiled, not run on hardware
//
#include <hip/hip_runtime.h>
#include <hip/hip_bf16.h>

typedef __attribute__((ext_vector_type(16))) _Float16 v16h;
typedef __attribute__((ext_vector_type(8)))  _Float16 v8h;
typedef __attribute__((ext_vector_type(8)))  float    v8f;
typedef long long i64;

#define DIM_IN 512
#define DIM_H1 128
#define DIM_H2 128
#define DIM_Z  64

static inline int cdiv(int a, int b) { return (a + b - 1) / b; }

// ---------------------------------------------------------------------------
// degree / symmetric norm
// ---------------------------------------------------------------------------
__global__ void init_deg_k(float* __restrict__ deg, int n) {
    int i = blockIdx.x * blockDim.x + threadIdx.x;
    if (i < n) deg[i] = 1.0f;   // self-loop contributes 1
}

__global__ void count_deg_k(const i64* __restrict__ dst, float* __restrict__ deg, int e) {
    int i = blockIdx.x * blockDim.x + threadIdx.x;
    if (i < e) atomicAdd(&deg[(int)dst[i]], 1.0f);
}

__global__ void dis_k(float* __restrict__ deg, int n) {
    int i = blockIdx.x * blockDim.x + threadIdx.x;
    if (i < n) deg[i] = rsqrtf(fmaxf(deg[i], 1.0f));
}

// ---------------------------------------------------------------------------
// Wt[n*kdim + k] = (f16) W[k*ndim + n]   (transpose + convert, done once)
// ---------------------------------------------------------------------------
__global__ void wt_k(const float* __restrict__ W, _Float16* __restrict__ Wt,
                     int kdim, int ndim)
{
    int idx = blockIdx.x * blockDim.x + threadIdx.x;
    if (idx >= kdim * ndim) return;
    int nn = idx / kdim, kk = idx - nn * kdim;
    Wt[idx] = (_Float16)W[(size_t)kk * ndim + nn];
}

// ---------------------------------------------------------------------------
// Fused GCN transform:
//   Y    = (RELU_IN ? relu(X) : X) @ W          (raw, for edge gather)
//   Yagg = bias + Y * dis^2                     (scatter-add init, self-loop)
// Block: 256 threads = 8 waves; ROWS=(8/(NDIM/16))*16 rows of A staged in LDS
// as f16. Per wave, per K=32 step: 2x ds_load_b128 (A) + 2x global b128 (B,
// transposed f16) + 1x v_wmma_f32_16x16x32_f16.
// In-place Yagg==X is safe: each block stages its own rows before the barrier.
// ---------------------------------------------------------------------------
template<int KDIM, int NDIM, bool RELU_IN>
__global__ void __launch_bounds__(256)
gemm_wmma_k(const float* __restrict__ X, const _Float16* __restrict__ Wt,
            const float* __restrict__ dis, const float* __restrict__ bias,
            float* __restrict__ Y, float* __restrict__ Yagg, int nrows)
{
    constexpr int TN   = NDIM / 16;     // column tiles per row-tile
    constexpr int RPB  = 8 / TN;        // row tiles per block
    constexpr int ROWS = RPB * 16;
    __shared__ _Float16 As[ROWS * KDIM];

    const int tid  = threadIdx.x;
    const int row0 = blockIdx.x * ROWS;

    // cooperative stage: f32 -> f16 row block into LDS (optional fused relu)
    for (int idx = tid; idx < ROWS * KDIM; idx += 256) {
        const int r = idx / KDIM;
        const int k = idx - r * KDIM;
        int srow = row0 + r;
        if (srow >= nrows) srow = nrows - 1;
        float v = X[(size_t)srow * KDIM + k];
        if (RELU_IN) v = fmaxf(v, 0.0f);
        As[idx] = (_Float16)v;
    }
    __syncthreads();

    const int lane   = tid & 31;
    const int wave   = tid >> 5;
    const int tile_n = wave % TN;
    const int tmloc  = wave / TN;
    const int tile_m = blockIdx.x * RPB + tmloc;
    if (tile_m * 16 >= nrows) return;     // wave-uniform exit (after barrier)

    const int arow = tmloc * 16 + (lane & 15);
    const int ncol = tile_n * 16 + (lane & 15);
    const int kh   = (lane >> 4) << 3;    // 0 or 8  (A half-group)
    const int kg   = (lane >> 4) << 4;    // 0 or 16 (unused)

    const _Float16* __restrict__ alds = As + arow * KDIM;
    const _Float16* __restrict__ wcol = Wt + (size_t)ncol * KDIM;

    v8f acc = {};
#pragma unroll 4
    for (int k0 = 0; k0 < KDIM; k0 += 32) {
        const v8h alo = *(const v8h*)(alds + k0 + kh);
        const v8h ahi = *(const v8h*)(alds + k0 + 16 + kh);
        const v8h blo = *(const v8h*)(wcol + k0 + kh);
        const v8h bhi = *(const v8h*)(wcol + k0 + 16 + kh);
        const v16h a = __builtin_shufflevector(alo, ahi, 0,1,2,3,4,5,6,7,8,9,10,11,12,13,14,15);
        const v16h b = __builtin_shufflevector(blo, bhi, 0,1,2,3,4,5,6,7,8,9,10,11,12,13,14,15);
        acc = __builtin_amdgcn_wmma_f32_16x16x32_f16(
            false, a, false, b, (short)0, acc, false, false);
    }

    // C/D layout: vgpr r -> M = rbase + r; valid tile => all rows valid
    const int rbase = tile_m * 16 + ((lane >> 4) << 3);
    float* __restrict__ yout = Y    + (size_t)rbase * NDIM + ncol;
    float* __restrict__ yagg = Yagg + (size_t)rbase * NDIM + ncol;
    const float bcol = bias[ncol];
#pragma unroll
    for (int r = 0; r < 8; ++r) {
        const float v = acc[r];
        const float d = dis[rbase + r];
        yout[r * NDIM] = v;
        yagg[r * NDIM] = bcol + v * d * d;
    }
}

// ---------------------------------------------------------------------------
// out[dst] += h[src] * dis[src]*dis[dst]   — (F/4) threads/edge, float4 gather
// ---------------------------------------------------------------------------
template<int F>
__global__ void scatter_k(const i64* __restrict__ src, const i64* __restrict__ dst,
                          const float* __restrict__ dis, const float* __restrict__ h,
                          float* __restrict__ out, int e)
{
    int gid = blockIdx.x * blockDim.x + threadIdx.x;
    constexpr int LPE = F >> 2;
    const int eidx = gid / LPE;
    if (eidx >= e) return;
    const int f = (gid - eidx * LPE) << 2;

    const int s = (int)src[eidx];
    const int d = (int)dst[eidx];
    const float nrm = dis[s] * dis[d];

    const float4 hv = *(const float4*)(h + (size_t)s * F + f);
    float* o = out + (size_t)d * F + f;
    atomicAdd(o + 0, hv.x * nrm);
    atomicAdd(o + 1, hv.y * nrm);
    atomicAdd(o + 2, hv.z * nrm);
    atomicAdd(o + 3, hv.w * nrm);
}

// ---------------------------------------------------------------------------
extern "C" void kernel_launch(void* const* d_in, const int* in_sizes, int n_in,
                              void* d_out, int out_size, void* d_ws, size_t ws_size,
                              hipStream_t stream)
{
    const float* x    = (const float*)d_in[0];
    const i64*   eix  = (const i64*)  d_in[1];
    const float* W1   = (const float*)d_in[2];
    const float* b1   = (const float*)d_in[3];
    const float* W2   = (const float*)d_in[4];
    const float* b2   = (const float*)d_in[5];
    const float* Wm   = (const float*)d_in[6];
    const float* bm   = (const float*)d_in[7];
    const float* Wsd  = (const float*)d_in[8];
    const float* bsd  = (const float*)d_in[9];

    const int n = in_sizes[0] / DIM_IN;
    const int e = in_sizes[1] / 2;
    const i64* src = eix;
    const i64* dst = eix + e;

    // workspace layout (all fully written before read)
    float* dis  = (float*)d_ws;
    float* hbuf = dis  + ((n + 3) & ~3);            // N x 128 (raw transform)
    float* abuf = hbuf + (size_t)n * DIM_H1;        // N x 128 (aggregated)
    float* hm   = abuf + (size_t)n * DIM_H1;        // N x 64
    float* hs   = hm   + (size_t)n * DIM_Z;         // N x 64
    _Float16* wt1 = (_Float16*)(hs + (size_t)n * DIM_Z);   // 512x128 (n-major)
    _Float16* wt2 = wt1 + DIM_IN * DIM_H1;                 // 128x128
    _Float16* wtm = wt2 + DIM_H1 * DIM_H2;                 // 128x64
    _Float16* wts = wtm + DIM_H2 * DIM_Z;                  // 128x64
    float* zm   = (float*)d_out;                    // N x 64
    float* zs   = zm + (size_t)n * DIM_Z;           // N x 64

    const int TB = 256;

    // ---- weights: transpose + f16 convert (tiny) ----
    wt_k<<<cdiv(DIM_IN * DIM_H1, TB), TB, 0, stream>>>(W1,  wt1, DIM_IN, DIM_H1);
    wt_k<<<cdiv(DIM_H1 * DIM_H2, TB), TB, 0, stream>>>(W2,  wt2, DIM_H1, DIM_H2);
    wt_k<<<cdiv(DIM_H2 * DIM_Z,  TB), TB, 0, stream>>>(Wm,  wtm, DIM_H2, DIM_Z);
    wt_k<<<cdiv(DIM_H2 * DIM_Z,  TB), TB, 0, stream>>>(Wsd, wts, DIM_H2, DIM_Z);

    // ---- degree / symmetric norm (before gemm1: epilogue uses dis) ----
    init_deg_k <<<cdiv(n, TB), TB, 0, stream>>>(dis, n);
    count_deg_k<<<cdiv(e, TB), TB, 0, stream>>>(dst, dis, e);
    dis_k      <<<cdiv(n, TB), TB, 0, stream>>>(dis, n);

    // ---- layer 1: abuf = b1 + selfloop(x@W1); hbuf = x@W1; scatter edges ----
    gemm_wmma_k<DIM_IN, DIM_H1, false>
        <<<cdiv(n, 16), TB, 0, stream>>>(x, wt1, dis, b1, hbuf, abuf, n);
    scatter_k<DIM_H1>
        <<<cdiv(e * (DIM_H1 / 4), TB), TB, 0, stream>>>(src, dst, dis, hbuf, abuf, e);

    // ---- layer 2: relu fused at staging; agg-init written in place ----
    gemm_wmma_k<DIM_H1, DIM_H2, true>
        <<<cdiv(n, 16), TB, 0, stream>>>(abuf, wt2, dis, b2, hbuf, abuf, n);
    scatter_k<DIM_H2>
        <<<cdiv(e * (DIM_H2 / 4), TB), TB, 0, stream>>>(src, dst, dis, hbuf, abuf, e);

    // ---- heads: relu fused at staging; agg-init straight into d_out ----
    gemm_wmma_k<DIM_H2, DIM_Z, true>
        <<<cdiv(n, 32), TB, 0, stream>>>(abuf, wtm, dis, bm, hm, zm, n);
    gemm_wmma_k<DIM_H2, DIM_Z, true>
        <<<cdiv(n, 32), TB, 0, stream>>>(abuf, wts, dis, bsd, hs, zs, n);

    scatter_k<DIM_Z>
        <<<cdiv(e * (DIM_Z / 4), TB), TB, 0, stream>>>(src, dst, dis, hm, zm, e);
    scatter_k<DIM_Z>
        <<<cdiv(e * (DIM_Z / 4), TB), TB, 0, stream>>>(src, dst, dis, hs, zs, e);
}
